// Critic_71777493451339
// MI455X (gfx1250) — matrix operand order repaired and checked by reading, not compiled, not run on hardware
//
#include <hip/hip_runtime.h>
#include <hip/hip_bf16.h>
#include <math.h>

// ---------------------------------------------------------------------------
// Types for WMMA fragments (CDNA5 / gfx1250, wave32)
// ---------------------------------------------------------------------------
typedef __attribute__((ext_vector_type(16))) _Float16 v16h;
typedef __attribute__((ext_vector_type(8)))  _Float16 v8h;
typedef __attribute__((ext_vector_type(8)))  float    v8f;

#define T_STEPS 512
#define BSZ     256
#define OBS     64
#define HID     512
#define FOURH   2048
#define KC      576          // concat K = HID + OBS
#define NROW    131072       // T*B
#define M1      512
#define M2      256

#define RGRID   32           // persistent LSTM grid (one WG per 64 gate columns)
#define RTHR    256          // 8 waves per WG
#define KCHUNK  192          // K staging chunk (576 = 3*192)
#define BSTRIDE 200          // padded LDS stride (halves) for staged B slab

static __device__ inline v16h make16(v8h lo, v8h hi) {
  return __builtin_shufflevector(lo, hi, 0,1,2,3,4,5,6,7,8,9,10,11,12,13,14,15);
}

// A-matrix fragment: 16x32 f16, row-major source with element stride `stride`.
// ISA layout: lanes 0-15 -> row=lane, K {kk+0..7, kk+16..23};
//             lanes 16-31 -> row=lane-16, K {kk+8..15, kk+24..31}.
static __device__ inline v16h load_a_frag(const _Float16* base, int stride,
                                          int m0, int kk, int lane) {
  int row = m0 + (lane & 15);
  int kb  = kk + ((lane & 16) ? 8 : 0);
  const _Float16* p = base + (size_t)row * stride + kb;
  v8h lo = *(const v8h*)(p);
  v8h hi = *(const v8h*)(p + 16);
  return make16(lo, hi);
}

// B-matrix fragment: 32x16 f16 where B[k][n] = W[n][k], W row-major [N,K].
// ISA layout: lane n in 0-15 holds K {kk+0..7, kk+16..23} of col n; lanes 16-31 hold K {kk+8..15, kk+24..31}.
// -> two 8-half runs of weight row `ncol` at kb and kb+16.
static __device__ inline v16h load_b_frag(const _Float16* w, int stride,
                                          int n0, int kk, int lane) {
  int ncol = n0 + (lane & 15);
  int kb   = kk + ((lane & 16) ? 8 : 0);
  const _Float16* p = w + (size_t)ncol * stride + kb;
  v8h lo = *(const v8h*)(p);
  v8h hi = *(const v8h*)(p + 16);
  return make16(lo, hi);
}

// Async global -> LDS copy of 16 bytes (ASYNCcnt-tracked, CDNA5 TDM-lite path).
static __device__ inline void async_copy_b128(const void* gptr, void* ldsptr) {
  unsigned lds = (unsigned)(uintptr_t)ldsptr;  // low 32 bits of LDS aperture addr
  asm volatile("global_load_async_to_lds_b128 %0, %1, off"
               :: "v"(lds), "v"(gptr) : "memory");
}
static __device__ inline void async_wait0() {
  asm volatile("s_wait_asynccnt 0x0" ::: "memory");
}

static __device__ inline float sigm(float x) { return 1.0f / (1.0f + expf(-x)); }

// Simple monotonic-phase grid barrier (counter reset each launch by prep kernel).
static __device__ inline void grid_sync(int* cnt, int target) {
  __syncthreads();
  if (threadIdx.x == 0) {
    __threadfence();
    atomicAdd(cnt, 1);
    while (__hip_atomic_load(cnt, __ATOMIC_ACQUIRE, __HIP_MEMORY_SCOPE_AGENT) < target)
      __builtin_amdgcn_s_sleep(1);
  }
  __syncthreads();
  __threadfence();
}

// ---------------------------------------------------------------------------
// Kernel 0: convert weights to fp16, combine biases, init recurrent state.
// ---------------------------------------------------------------------------
__global__ void prep_kernel(const float* __restrict__ x, const float* __restrict__ done,
                            const float* __restrict__ h0, const float* __restrict__ c0,
                            const float* __restrict__ W_ih, const float* __restrict__ W_hh,
                            const float* __restrict__ b_ih, const float* __restrict__ b_hh,
                            const float* __restrict__ W1,  const float* __restrict__ W2,
                            _Float16* __restrict__ Wcat, _Float16* __restrict__ W1h,
                            _Float16* __restrict__ W2h, float* __restrict__ bg,
                            _Float16* __restrict__ ha, float* __restrict__ cbuf,
                            int* __restrict__ cnt) {
  const int tid = blockIdx.x * blockDim.x + threadIdx.x;
  const int nth = gridDim.x * blockDim.x;
  if (tid == 0) *cnt = 0;
  // Wcat[n][0:512] = W_hh[n], Wcat[n][512:576] = W_ih[n]
  for (int i = tid; i < FOURH * KC; i += nth) {
    int n = i / KC, k = i - n * KC;
    float v = (k < HID) ? W_hh[(size_t)n * HID + k] : W_ih[(size_t)n * OBS + (k - HID)];
    Wcat[i] = (_Float16)v;
  }
  for (int i = tid; i < M1 * HID; i += nth) W1h[i] = (_Float16)W1[i];
  for (int i = tid; i < M2 * M1;  i += nth) W2h[i] = (_Float16)W2[i];
  for (int i = tid; i < FOURH;    i += nth) bg[i]  = b_ih[i] + b_hh[i];
  // ha for t=0: [h0 * (1-done_0) ; x_0]
  for (int i = tid; i < BSZ * KC; i += nth) {
    int b = i / KC, k = i - b * KC;
    float v = (k < HID) ? h0[(size_t)b * HID + k] * (1.0f - done[b])
                        : x[(size_t)b * OBS + (k - HID)];
    ha[i] = (_Float16)v;
  }
  for (int i = tid; i < BSZ * HID; i += nth) cbuf[i] = c0[i];
}

// ---------------------------------------------------------------------------
// Kernel 1: persistent LSTM recurrence.
// 32 WGs; WG owns 64 gate columns (4 nt tiles) x all 16 batch tiles = 64 tiles.
// B slab staged once per WG per K-chunk via async global->LDS (unique-weight
// L2 traffic: 2.3 MB/step total); waves read B fragments from LDS (ds_load),
// A fragments from the small L2-resident [h;x] activation buffer.
// Each wave: 2 batch tiles x 4 column tiles = 8 f32 accumulators.
// ---------------------------------------------------------------------------
__global__ void __launch_bounds__(RTHR)
lstm_kernel(const float* __restrict__ x, const float* __restrict__ done,
            const float* __restrict__ bg, const _Float16* __restrict__ Wcat,
            _Float16* __restrict__ ha, float* __restrict__ cbuf,
            float* __restrict__ gates, _Float16* __restrict__ hs16,
            float* __restrict__ outHT, float* __restrict__ outCT,
            int* __restrict__ cnt) {
  __shared__ __align__(32) _Float16 Bs[64 * BSTRIDE];   // 25.6 KB staged weights

  const int lane = threadIdx.x & 31;
  const int wave = threadIdx.x >> 5;
  const int tid  = blockIdx.x * RTHR + threadIdx.x;
  const int col0 = blockIdx.x * 64;                     // first gate column of WG
  const int bt0  = wave * 2;                            // this wave's batch tiles
  int phase = 0;

  for (int t = 0; t < T_STEPS; ++t) {
    grid_sync(cnt, (++phase) * RGRID);                  // ha for step t ready

    v8f acc[8] = {};                                    // [ib*4 + jn]
    #pragma unroll 1
    for (int kc = 0; kc < KC; kc += KCHUNK) {
      // ---- stage B slab: 64 cols x 192 K halves = 1536 x 16B transfers ----
      #pragma unroll
      for (int j = 0; j < 6; ++j) {
        int idx = threadIdx.x + 256 * j;                // 0..1535
        int row = idx / 24;                             // local column 0..63
        int seg = idx - row * 24;                       // 16B segment 0..23
        const _Float16* g = Wcat + (size_t)(col0 + row) * KC + kc + seg * 8;
        async_copy_b128(g, &Bs[row * BSTRIDE + seg * 8]);
      }
      async_wait0();
      __syncthreads();

      // ---- compute on staged chunk ----
      #pragma unroll 1
      for (int kk = 0; kk < KCHUNK; kk += 32) {
        v16h a0 = load_a_frag(ha, KC, (bt0 + 0) * 16, kc + kk, lane);
        v16h a1 = load_a_frag(ha, KC, (bt0 + 1) * 16, kc + kk, lane);
        #pragma unroll
        for (int jn = 0; jn < 4; ++jn) {
          v16h b = load_b_frag((const _Float16*)Bs, BSTRIDE, jn * 16, kk, lane);
          acc[0 * 4 + jn] = __builtin_amdgcn_wmma_f32_16x16x32_f16(
              false, a0, false, b, (short)0, acc[0 * 4 + jn], false, false);
          acc[1 * 4 + jn] = __builtin_amdgcn_wmma_f32_16x16x32_f16(
              false, a1, false, b, (short)0, acc[1 * 4 + jn], false, false);
        }
      }
      __syncthreads();                                  // protect Bs reuse
    }

    // ---- store gate tiles ----
    #pragma unroll
    for (int ib = 0; ib < 2; ++ib) {
      #pragma unroll
      for (int jn = 0; jn < 4; ++jn) {
        int ncol = col0 + jn * 16 + (lane & 15);
        int mr0  = (bt0 + ib) * 16 + ((lane & 16) ? 8 : 0);
        float* gp = gates + (size_t)mr0 * FOURH + ncol;
        #pragma unroll
        for (int r = 0; r < 8; ++r) gp[(size_t)r * FOURH] = acc[ib * 4 + jn][r];
      }
    }

    grid_sync(cnt, (++phase) * RGRID);                  // gates ready

    // ---- cell update + assemble ha for t+1 ----
    for (int e = tid; e < BSZ * HID; e += RGRID * RTHR) {
      int b = e >> 9, k = e & (HID - 1);
      const float* gr = gates + (size_t)b * FOURH;
      float gi = gr[k]           + bg[k];
      float gf = gr[HID + k]     + bg[HID + k];
      float gg = gr[2 * HID + k] + bg[2 * HID + k];
      float go = gr[3 * HID + k] + bg[3 * HID + k];
      float m  = 1.0f - done[t * BSZ + b];
      float cn = sigm(gf) * (cbuf[e] * m) + sigm(gi) * tanhf(gg);
      float hn = sigm(go) * tanhf(cn);
      cbuf[e] = cn;
      hs16[(size_t)(t * BSZ + b) * HID + k] = (_Float16)hn;
      if (t + 1 < T_STEPS) {
        float m2 = 1.0f - done[(t + 1) * BSZ + b];
        ha[(size_t)b * KC + k] = (_Float16)(hn * m2);
      } else {
        outHT[e] = hn;
        outCT[e] = cn;
      }
    }
    if (t + 1 < T_STEPS) {
      for (int e = tid; e < BSZ * OBS; e += RGRID * RTHR) {
        int b = e >> 6, j = e & (OBS - 1);
        ha[(size_t)b * KC + HID + j] =
            (_Float16)x[((size_t)(t + 1) * BSZ + b) * OBS + j];
      }
    }
  }
}

// ---------------------------------------------------------------------------
// Kernel 2: fused LayerNorm + tanh-MLP + value head. One WG per 16 rows.
// Hidden tile arrives via async global->LDS; both GEMM layers use WMMA with
// A-fragments from LDS, weights from L2.
// ---------------------------------------------------------------------------
#define LNS 520   // padded LDS row stride (halves), keeps 16B alignment
__global__ void __launch_bounds__(256)
mlp_kernel(const _Float16* __restrict__ hs16,
           const float* __restrict__ ln_g, const float* __restrict__ ln_b,
           const _Float16* __restrict__ W1h, const float* __restrict__ b1,
           const _Float16* __restrict__ W2h, const float* __restrict__ b2,
           const float* __restrict__ Wv, const float* __restrict__ bv,
           float* __restrict__ value) {
  __shared__ __align__(32) _Float16 hln[16][LNS];
  __shared__ __align__(32) _Float16 h1 [16][LNS];
  __shared__ __align__(16) float    h2 [16][264];

  const int lane = threadIdx.x & 31;
  const int wave = threadIdx.x >> 5;
  const size_t r0 = (size_t)blockIdx.x * 16;

  // async-load 16 x 512 fp16 tile of hidden states into LDS (1024 x 16B)
  #pragma unroll
  for (int j = 0; j < 4; ++j) {
    int idx = threadIdx.x + 256 * j;                 // 0..1023
    int r   = idx >> 6;                              // row 0..15
    int seg = idx & 63;                              // 16B segment 0..63
    async_copy_b128(hs16 + (r0 + r) * HID + seg * 8, &hln[r][seg * 8]);
  }
  async_wait0();
  __syncthreads();

  // LayerNorm: each wave owns 2 rows
  #pragma unroll
  for (int rr = 0; rr < 2; ++rr) {
    int r = wave * 2 + rr;
    float s = 0.0f;
    for (int k = lane; k < HID; k += 32) s += (float)hln[r][k];
    for (int off = 16; off; off >>= 1) s += __shfl_xor(s, off, 32);
    float mu = s * (1.0f / HID);
    float v = 0.0f;
    for (int k = lane; k < HID; k += 32) {
      float d = (float)hln[r][k] - mu; v += d * d;
    }
    for (int off = 16; off; off >>= 1) v += __shfl_xor(v, off, 32);
    float rs = rsqrtf(v * (1.0f / HID) + 1e-5f);
    for (int k = lane; k < HID; k += 32) {
      float d = ((float)hln[r][k] - mu) * rs * ln_g[k] + ln_b[k];
      hln[r][k] = (_Float16)d;
    }
  }
  __syncthreads();

  // Layer 1: 32 column tiles, 4 per wave; tanh epilogue into LDS (fp16)
  #pragma unroll 1
  for (int i = 0; i < 4; ++i) {
    int ct = wave * 4 + i;
    __builtin_prefetch(W1h + (size_t)(ct * 16) * HID, 0, 0);
    v8f acc = {};
    #pragma unroll
    for (int kk = 0; kk < HID; kk += 32) {
      v16h a = load_a_frag(&hln[0][0], LNS, 0, kk, lane);
      v16h b = load_b_frag(W1h, HID, ct * 16, kk, lane);
      acc = __builtin_amdgcn_wmma_f32_16x16x32_f16(
          false, a, false, b, (short)0, acc, false, false);
    }
    int ncol = ct * 16 + (lane & 15);
    int m0   = (lane & 16) ? 8 : 0;
    float bb = b1[ncol];
    #pragma unroll
    for (int r = 0; r < 8; ++r) h1[m0 + r][ncol] = (_Float16)tanhf(acc[r] + bb);
  }
  __syncthreads();

  // Layer 2: 16 column tiles, 2 per wave; tanh epilogue into LDS (fp32)
  #pragma unroll 1
  for (int i = 0; i < 2; ++i) {
    int ct = wave * 2 + i;
    v8f acc = {};
    #pragma unroll
    for (int kk = 0; kk < M1; kk += 32) {
      v16h a = load_a_frag(&h1[0][0], LNS, 0, kk, lane);
      v16h b = load_b_frag(W2h, M1, ct * 16, kk, lane);
      acc = __builtin_amdgcn_wmma_f32_16x16x32_f16(
          false, a, false, b, (short)0, acc, false, false);
    }
    int ncol = ct * 16 + (lane & 15);
    int m0   = (lane & 16) ? 8 : 0;
    float bb = b2[ncol];
    #pragma unroll
    for (int r = 0; r < 8; ++r) h2[m0 + r][ncol] = tanhf(acc[r] + bb);
  }
  __syncthreads();

  // Value head: dot(h2[row], Wv) + bv ; each wave owns 2 rows
  #pragma unroll
  for (int rr = 0; rr < 2; ++rr) {
    int r = wave * 2 + rr;
    float s = 0.0f;
    for (int k = lane; k < M2; k += 32) s += h2[r][k] * Wv[k];
    for (int off = 16; off; off >>= 1) s += __shfl_xor(s, off, 32);
    if (lane == 0) value[r0 + r] = s + bv[0];
  }
}

// ---------------------------------------------------------------------------
// Host launcher
// ---------------------------------------------------------------------------
extern "C" void kernel_launch(void* const* d_in, const int* in_sizes, int n_in,
                              void* d_out, int out_size, void* d_ws, size_t ws_size,
                              hipStream_t stream) {
  const float* x    = (const float*)d_in[0];
  const float* done = (const float*)d_in[1];
  const float* h0   = (const float*)d_in[2];
  const float* c0   = (const float*)d_in[3];
  const float* W_ih = (const float*)d_in[4];
  const float* W_hh = (const float*)d_in[5];
  const float* b_ih = (const float*)d_in[6];
  const float* b_hh = (const float*)d_in[7];
  const float* ln_g = (const float*)d_in[8];
  const float* ln_b = (const float*)d_in[9];
  const float* W1   = (const float*)d_in[10];
  const float* b1   = (const float*)d_in[11];
  const float* W2   = (const float*)d_in[12];
  const float* b2   = (const float*)d_in[13];
  const float* Wv   = (const float*)d_in[14];
  const float* bv   = (const float*)d_in[15];

  char* ws = (char*)d_ws;
  size_t off = 0;
  auto take = [&](size_t bytes) -> char* {
    char* p = ws + off;
    off = (off + bytes + 255) & ~(size_t)255;
    return p;
  };
  int*      cnt   = (int*)take(256);
  _Float16* Wcat  = (_Float16*)take((size_t)FOURH * KC * 2);
  _Float16* W1h   = (_Float16*)take((size_t)M1 * HID * 2);
  _Float16* W2h   = (_Float16*)take((size_t)M2 * M1 * 2);
  float*    bg    = (float*)take((size_t)FOURH * 4);
  _Float16* ha    = (_Float16*)take((size_t)BSZ * KC * 2);
  float*    cbuf  = (float*)take((size_t)BSZ * HID * 4);
  float*    gates = (float*)take((size_t)BSZ * FOURH * 4);
  _Float16* hs16  = (_Float16*)take((size_t)NROW * HID * 2);

  float* value = (float*)d_out;          // [T*B]
  float* outHT = value + NROW;           // [B*H]
  float* outCT = outHT + BSZ * HID;      // [B*H]

  prep_kernel<<<512, 256, 0, stream>>>(x, done, h0, c0, W_ih, W_hh, b_ih, b_hh,
                                       W1, W2, Wcat, W1h, W2h, bg, ha, cbuf, cnt);
  lstm_kernel<<<RGRID, RTHR, 0, stream>>>(x, done, bg, Wcat, ha, cbuf, gates,
                                          hs16, outHT, outCT, cnt);
  mlp_kernel<<<NROW / 16, 256, 0, stream>>>(hs16, ln_g, ln_b, W1h, b1, W2h, b2,
                                            Wv, bv, value);
}
